// MlpwithSOMCosSimilarityModule_71708773974142
// MI455X (gfx1250) — matrix operand order repaired and checked by reading, not compiled, not run on hardware
//
#include <hip/hip_runtime.h>
#include <math.h>

typedef __attribute__((ext_vector_type(16))) __bf16 v16bf;
typedef __attribute__((ext_vector_type(8)))  float  v8f;

#define LDIM 128
#define EDIM 768
#define CHK  32
#define NCH  (EDIM / CHK)   // 24 K-chunks of 32
#define RS   40             // LDS row stride in halves: 32 data + 8 pad (80B, 16B aligned)

__device__ __forceinline__ unsigned short f2bf(float f) {
    unsigned u = __float_as_uint(f);
    unsigned r = u + 0x7FFFu + ((u >> 16) & 1u);   // round-to-nearest-even
    return (unsigned short)(r >> 16);
}

union FragCvt { uint4 q[2]; v16bf v; };

__global__ __launch_bounds__(256)
void som_cossim_mlp_kernel(const float* __restrict__ ctx,
                           const float* __restrict__ w0, const float* __restrict__ b0,
                           const float* __restrict__ w1, const float* __restrict__ b1,
                           const float* __restrict__ w2, const float* __restrict__ b2,
                           const float* __restrict__ w3, const float* __restrict__ b3,
                           const float* __restrict__ w4, const float* __restrict__ b4,
                           const float* __restrict__ w5, const float* __restrict__ b5,
                           const float* __restrict__ w6, const float* __restrict__ b6,
                           float* __restrict__ out)
{
    __shared__ __align__(16) unsigned short sC[LDIM * RS];   // context chunk, bf16
    __shared__ __align__(16) unsigned short sE[LDIM * RS];   // entity  chunk, bf16
    __shared__ __align__(16) float sW0[5 * 2 * EDIM];        // full (5,1536) layer-0 weights
    __shared__ float sInvC[LDIM];
    __shared__ float sInvE[LDIM];
    __shared__ float sPc[LDIM][5];                           // W0c . c[m]
    __shared__ float sPe[LDIM][5];                           // W0e . e[n]
    __shared__ int   sIdx[LDIM];
    __shared__ float sRed[8];

    const int t    = threadIdx.x;
    const int bk   = blockIdx.x;          // 0..1023 : (b,k) batch
    const int lane = t & 31;
    const int wid  = t >> 5;
    const int lh   = lane & 15;
    const int hf   = lane >> 4;

    // Stage layer-0 weights into LDS (broadcast reads later, conflict-free)
    for (int i = t; i < 5 * 2 * EDIM; i += 256) sW0[i] = w0[i];

    // Each thread streams one row: t<128 -> context row t ; else entity row t-128
    const bool  isC = (t < LDIM);
    const int   r   = isC ? t : t - LDIM;
    const float* __restrict__ rowBase =
        ctx + ((size_t)bk * 2 + (isC ? 0 : 1)) * (size_t)(LDIM * EDIM) + (size_t)r * EDIM;
    const int wb = isC ? 0 : EDIM;        // column offset into w0

    float sumsq = 0.f;
    float P[5]  = {0.f, 0.f, 0.f, 0.f, 0.f};

    v8f acc[8];
    #pragma unroll
    for (int i = 0; i < 8; i++)
        #pragma unroll
        for (int j = 0; j < 8; j++) acc[i][j] = 0.f;

    // Preload chunk 0
    float4 v[8];
    #pragma unroll
    for (int i = 0; i < 8; i++) v[i] = *(const float4*)(rowBase + 4 * i);

    __syncthreads();   // sW0 ready

    for (int ch = 0; ch < NCH; ++ch) {
        const int k0 = ch * CHK;

        // ---- consume chunk: sum-of-squares, layer-0 partials, bf16 pack ----
        unsigned pw[16];
        #pragma unroll
        for (int i = 0; i < 8; i++) {
            const float4 x = v[i];
            sumsq += x.x * x.x + x.y * x.y + x.z * x.z + x.w * x.w;
            #pragma unroll
            for (int j = 0; j < 5; j++) {
                const float4 w = *(const float4*)&sW0[j * (2 * EDIM) + wb + k0 + 4 * i];
                P[j] += w.x * x.x + w.y * x.y + w.z * x.z + w.w * x.w;
            }
            pw[2 * i]     = (unsigned)f2bf(x.x) | ((unsigned)f2bf(x.y) << 16);
            pw[2 * i + 1] = (unsigned)f2bf(x.z) | ((unsigned)f2bf(x.w) << 16);
        }

        __syncthreads();   // prior WMMA reads of sC/sE complete
        {
            uint4* dst = (uint4*)((isC ? sC : sE) + (size_t)r * RS);
            dst[0] = *(const uint4*)&pw[0];
            dst[1] = *(const uint4*)&pw[4];
            dst[2] = *(const uint4*)&pw[8];
            dst[3] = *(const uint4*)&pw[12];
        }
        __syncthreads();   // bf16 tiles visible

        // Issue next chunk's HBM loads early so they overlap the WMMAs below
        if (ch + 1 < NCH) {
            #pragma unroll
            for (int i = 0; i < 8; i++)
                v[i] = *(const float4*)(rowBase + k0 + CHK + 4 * i);
            __builtin_prefetch(rowBase + k0 + 2 * CHK, 0, 0);
        }

        // ---- WMMA: wave owns rows m0..m0+15, all 128 cols (8 tiles) ----
        const int m0 = wid * 16;
        // A frag (16x32 bf16): lane<16 -> row m0+lh K[0..7],[16..23]; lane>=16 -> K[8..15],[24..31]
        FragCvt fa;
        {
            const uint4* arow = (const uint4*)(sC + (size_t)(m0 + lh) * RS);
            fa.q[0] = arow[hf ? 1 : 0];
            fa.q[1] = arow[hf ? 3 : 2];
        }
        const v16bf afrag = fa.v;
        #pragma unroll
        for (int nt = 0; nt < 8; ++nt) {
            // B frag (32x16 bf16), B = E^T: lane<16 -> row 16*nt+lh K[0..7],[16..23]; lane>=16 -> K[8..15],[24..31]
            FragCvt fb;
            const uint4* brow = (const uint4*)(sE + (size_t)(nt * 16 + lh) * RS);
            fb.q[0] = brow[hf ? 1 : 0];
            fb.q[1] = brow[hf ? 3 : 2];
            acc[nt] = __builtin_amdgcn_wmma_f32_16x16x32_bf16(
                false, afrag, false, fb.v, (short)0, acc[nt], false, false);
        }
    }

    // ---- per-row norms + layer-0 partials to LDS ----
    {
        const float inv = rsqrtf(sumsq);
        if (isC) {
            sInvC[r] = inv;
            #pragma unroll
            for (int j = 0; j < 5; j++) sPc[r][j] = P[j];
        } else {
            sInvE[r] = inv;
            #pragma unroll
            for (int j = 0; j < 5; j++) sPe[r][j] = P[j];
        }
    }
    __syncthreads();

    // ---- argmax over n of dot[m][n] / |e_n|  (C/D layout: vgpr j, half hf -> row j+8*hf) ----
    {
        const int m0 = wid * 16;
        #pragma unroll
        for (int j = 0; j < 8; j++) {
            float bv = -3.402823e38f;
            int   bn = 0;
            #pragma unroll
            for (int nt = 0; nt < 8; nt++) {
                const int n = nt * 16 + lh;
                const float val = acc[nt][j] * sInvE[n];
                if (val > bv) { bv = val; bn = n; }   // first-occurrence tie-break (n ascending)
            }
            #pragma unroll
            for (int off = 1; off < 16; off <<= 1) {
                const float ov = __shfl_xor(bv, off, 32);
                const int   on = __shfl_xor(bn, off, 32);
                if (ov > bv || (ov == bv && on < bn)) { bv = ov; bn = on; }
            }
            if (lh == 0) sIdx[m0 + hf * 8 + j] = bn;
        }
    }
    __syncthreads();

    // ---- tiny MLP tail per row + reduction over L ----
    float y = 0.f;
    if (t < LDIM) {
        const int   m   = t;
        const int   n   = sIdx[m];
        const float icn = sInvC[m];
        const float ien = sInvE[n];
        float h0[5];
        #pragma unroll
        for (int j = 0; j < 5; j++)
            h0[j] = tanhf(sPc[m][j] * icn + sPe[n][j] * ien + b0[j]);
        float h1[2];
        #pragma unroll
        for (int j = 0; j < 2; j++) {
            float s = b1[j];
            #pragma unroll
            for (int q = 0; q < 5; q++) s += w1[j * 5 + q] * h0[q];
            h1[j] = tanhf(s);
        }
        const float h2 = tanhf(w2[0] * h1[0] + w2[1] * h1[1] + b2[0]);
        const float h3 = tanhf(w3[0] * h2 + b3[0]);
        const float h4 = tanhf(w4[0] * h3 + b4[0]);
        const float h5 = tanhf(w5[0] * h4 + b5[0]);
        y = w6[0] * h5 + b6[0];
    }
    #pragma unroll
    for (int off = 16; off; off >>= 1) y += __shfl_xor(y, off, 32);
    if (lane == 0) sRed[wid] = y;
    __syncthreads();
    if (t == 0) {
        float tot = 0.f;
        #pragma unroll
        for (int i = 0; i < 8; i++) tot += sRed[i];
        out[bk] = tot;
    }
}

extern "C" void kernel_launch(void* const* d_in, const int* in_sizes, int n_in,
                              void* d_out, int out_size, void* d_ws, size_t ws_size,
                              hipStream_t stream)
{
    (void)in_sizes; (void)n_in; (void)out_size; (void)d_ws; (void)ws_size;
    const float* ctx = (const float*)d_in[0];
    const float* w0  = (const float*)d_in[1];
    const float* b0  = (const float*)d_in[2];
    const float* w1  = (const float*)d_in[3];
    const float* b1  = (const float*)d_in[4];
    const float* w2  = (const float*)d_in[5];
    const float* b2  = (const float*)d_in[6];
    const float* w3  = (const float*)d_in[7];
    const float* b3  = (const float*)d_in[8];
    const float* w4  = (const float*)d_in[9];
    const float* b4  = (const float*)d_in[10];
    const float* w5  = (const float*)d_in[11];
    const float* b5  = (const float*)d_in[12];
    const float* w6  = (const float*)d_in[13];
    const float* b6  = (const float*)d_in[14];
    float* out = (float*)d_out;

    dim3 grid(1024), block(256);
    hipLaunchKernelGGL(som_cossim_mlp_kernel, grid, block, 0, stream,
                       ctx, w0, b0, w1, b1, w2, b2, w3, b3, w4, b4, w5, b5, w6, b6, out);
}
